// Model_25451976196694
// MI455X (gfx1250) — matrix operand order repaired and checked by reading, not compiled, not run on hardware
//
#include <hip/hip_runtime.h>
#include <hip/hip_bf16.h>
#include <math.h>
#include <stdint.h>

typedef __attribute__((ext_vector_type(8)))  _Float16 h8;
typedef __attribute__((ext_vector_type(16))) _Float16 v16h;
typedef __attribute__((ext_vector_type(8)))  float    v8f;

#define B_    8
#define N_    2048
#define D_    64
#define S_    8
#define R_    8
#define HD_   8
#define HV_   8
#define PSTR  2052                   // f32 stripe stride (2048+4) -> bank spread
#define PSTRH 2064                   // f16 stripe stride -> 16B-aligned, bank spread
#define SCALE_ 0.35355339059327379f  // 8^-0.5

union V16 { v16h v; h8 h[2]; _Float16 e[16]; };

// ---------------------------------------------------------------- helpers
__device__ __forceinline__ void gemv64(const float* __restrict__ v,
                                       const float* __restrict__ W,
                                       const float* __restrict__ bias,
                                       float* __restrict__ out) {
  #pragma unroll
  for (int j = 0; j < 64; ++j) out[j] = bias[j];
  #pragma unroll 4
  for (int i = 0; i < 64; ++i) {
    const float vi = v[i];
    const float* wr = W + i * 64;   // uniform address -> scalar loads
    #pragma unroll
    for (int j = 0; j < 64; ++j) out[j] = __builtin_fmaf(vi, wr[j], out[j]);
  }
}

// ================================================================ kernel 1
// encoder (Linear->ReLU->Linear) + q/k/v/qv projections.
// q,k stored f16 [B,S,N,8]; v stored f16 transposed [B,64,N]; qv f32 [B,N,64].
__global__ __launch_bounds__(256) void k_encode(
    const float* __restrict__ x,
    const float* __restrict__ w1, const float* __restrict__ b1,
    const float* __restrict__ w2, const float* __restrict__ b2,
    const float* __restrict__ wq, const float* __restrict__ bq,
    const float* __restrict__ wk, const float* __restrict__ bk,
    const float* __restrict__ wv, const float* __restrict__ bv,
    const float* __restrict__ wvq, const float* __restrict__ bvq,
    _Float16* __restrict__ q16, _Float16* __restrict__ k16,
    _Float16* __restrict__ vt, float* __restrict__ qvbuf) {
  const int idx = blockIdx.x * 256 + threadIdx.x;   // 0..16383
  const int b = idx >> 11, n = idx & (N_ - 1);
  const float* xr = x + (size_t)idx * D_;

  float h1[D_], h2[D_], acc[D_];
  gemv64(xr, w1, b1, h1);
  #pragma unroll
  for (int j = 0; j < D_; ++j) h1[j] = fmaxf(h1[j], 0.f);
  gemv64(h1, w2, b2, h2);

  gemv64(h2, wq, bq, acc);
  #pragma unroll
  for (int s = 0; s < S_; ++s)
    #pragma unroll
    for (int hd = 0; hd < HD_; ++hd)
      q16[((size_t)(b * S_ + s) * N_ + n) * HD_ + hd] =
          (_Float16)(acc[s * 8 + hd] * SCALE_);

  gemv64(h2, wk, bk, acc);
  #pragma unroll
  for (int s = 0; s < S_; ++s)
    #pragma unroll
    for (int hd = 0; hd < HD_; ++hd)
      k16[((size_t)(b * S_ + s) * N_ + n) * HD_ + hd] = (_Float16)acc[s * 8 + hd];

  gemv64(h2, wv, bv, acc);
  #pragma unroll
  for (int d = 0; d < D_; ++d)
    vt[((size_t)(b * 64 + d)) * N_ + n] = (_Float16)acc[d];

  gemv64(h2, wvq, bvq, acc);
  #pragma unroll
  for (int d = 0; d < D_; ++d)
    qvbuf[(size_t)idx * 64 + d] = acc[d] * SCALE_;
}

// ================================================================ kernel 2
// One WG = (b, search-head sh, 16-row stripe).
//  - async-stage K panel (32 KB) to LDS (ASYNCcnt path)
//  - QK (f16 WMMA) -> 128 KB LDS f32 stripe
//  - masked softmax in-place, single attn HBM write, f16 P staged to LDS
//  - PV (f16 WMMA, dual accumulators) with V streamed from L2
//  - fused stage-2 retrieval attention
__global__ __launch_bounds__(256) void k_attn(
    const _Float16* __restrict__ q16, const _Float16* __restrict__ k16,
    const _Float16* __restrict__ vt, const float* __restrict__ qvbuf,
    const float* __restrict__ gw, const float* __restrict__ gb,
    float* __restrict__ attn_out, float* __restrict__ vscore_out,
    float* __restrict__ obuf) {
  extern __shared__ float ldsP[];          // [16][PSTR]  f32 scores (128 KB)
  _Float16* ldsPH = (_Float16*)(ldsP + 16 * PSTR);  // [16][PSTRH] f16 P (64.5 KB)
  _Float16* ldsK  = ldsPH;                 // K panel aliases f16-P region (32 KB)
  __shared__ float outbuf[16 * 68];        // PV result [16 rows][64 d]
  __shared__ float lred[256];
  __shared__ float rstat[16];
  __shared__ float rinv[16];

  const int tid = threadIdx.x;
  const int lane = tid & 31, wave = tid >> 5;
  const int lm = lane & 15, lhalf = lane >> 4;
  const int id = blockIdx.x;
  const int nt = id & 127, sh = (id >> 7) & 7, b = id >> 10;
  const int n0 = nt << 4;

  const _Float16* qbase = q16 + (size_t)(b * S_ + sh) * N_ * HD_;
  const _Float16* kbase = k16 + (size_t)(b * S_ + sh) * N_ * HD_;

  // ---------- Phase 0: async-stage K panel [2048][8] f16 into LDS ----------
  {
    const uint32_t lbase = (uint32_t)(uintptr_t)ldsK;  // low 32b = LDS offset
    #pragma unroll
    for (int i = 0; i < 8; ++i) {
      const uint32_t ofs = (uint32_t)(tid + (i << 8)) << 4;   // 16 B / lane
      const uint64_t ga = (uint64_t)(uintptr_t)kbase + ofs;
      const uint32_t la = lbase + ofs;
      asm volatile("global_load_async_to_lds_b128 %0, %1, off"
                   :: "v"(la), "v"(ga) : "memory");
    }
  }

  for (int i = tid; i < 16 * 68; i += 256) outbuf[i] = 0.f;

  // A fragment: lane holds row M=lm, K0..7 in halves 0..7 (K padded 8->32)
  V16 a, bb;
  #pragma unroll
  for (int i = 0; i < 16; ++i) { a.e[i] = (_Float16)0.f; bb.e[i] = (_Float16)0.f; }
  if (lhalf == 0)
    a.h[0] = *(const h8*)(qbase + (size_t)(n0 + lm) * HD_);

  asm volatile("s_wait_asynccnt 0x0" ::: "memory");
  __syncthreads();

  // ---------- Phase 1: scores = Q K^T (K panel now in LDS) ----------
  for (int i = 0; i < 16; ++i) {
    const int mt = wave + (i << 3);
    const int m0 = mt << 4;
    if (lhalf == 0)  // B: lane holds column N=lm, K0..7 contiguous (ds_load_b128)
      bb.h[0] = *(const h8*)(ldsK + (size_t)(m0 + lm) * HD_);
    v8f c = {0.f, 0.f, 0.f, 0.f, 0.f, 0.f, 0.f, 0.f};
    c = __builtin_amdgcn_wmma_f32_16x16x32_f16(false, a.v, false, bb.v,
                                               (short)0, c, false, false);
    if (m0 == n0) {  // diagonal exclusion
      #pragma unroll
      for (int r = 0; r < 8; ++r)
        if ((r + (lhalf << 3)) == lm) c[r] = -3.0e38f;
    }
    #pragma unroll
    for (int r = 0; r < 8; ++r)  // C/D layout: M=r+8*lhalf, Ncol=lm
      ldsP[(r + (lhalf << 3)) * PSTR + m0 + lm] = c[r];
  }
  __syncthreads();

  // ---------- Phase 2: softmax in LDS, write attn once, stage f16 P ----------
  {
    const int row = tid >> 4, seg = tid & 15;
    float m = -3.4e38f;
    for (int i = 0; i < 128; ++i)
      m = fmaxf(m, ldsP[row * PSTR + seg + (i << 4)]);
    lred[tid] = m;
    __syncthreads();
    if (seg == 0) {
      float mm = lred[tid];
      #pragma unroll
      for (int j = 1; j < 16; ++j) mm = fmaxf(mm, lred[tid + j]);
      rstat[row] = mm;
    }
    __syncthreads();
    const float rm = rstat[row];
    float ssum = 0.f;
    for (int i = 0; i < 128; ++i) {
      const int o = row * PSTR + seg + (i << 4);
      const float e = __expf(ldsP[o] - rm);
      ldsP[o] = e;
      ssum += e;
    }
    lred[tid] = ssum;
    __syncthreads();
    if (seg == 0) {
      float s2 = lred[tid];
      #pragma unroll
      for (int j = 1; j < 16; ++j) s2 += lred[tid + j];
      rinv[row] = 1.f / s2;
    }
    __syncthreads();
    const float inv = rinv[row];
    const size_t gbase =
        ((size_t)(b * S_ + sh) * N_ + (size_t)(n0 + row)) * N_;
    for (int i = 0; i < 128; ++i) {
      const int c2 = seg + (i << 4);
      const float p = ldsP[row * PSTR + c2] * inv;
      ldsPH[row * PSTRH + c2] = (_Float16)p;  // f16 P for the WMMA A-path
      attn_out[gbase + c2] = p;               // single mandatory HBM write
    }
  }
  __syncthreads();

  // ---------- Phase 3: out = P @ V (f16 WMMA, V streamed from L2) ----------
  {
    const int dt = wave >> 1, half = wave & 1;
    const int d = (dt << 4) + lm;                 // B: lane holds column d
    const _Float16* vrow =
        vt + ((size_t)(b * 64 + d)) * N_ + (lhalf << 4);
    const int cofs = lhalf << 4;                  // A K-offset per lane half
    const _Float16* arow = ldsPH + lm * PSTRH + cofs;
    v8f acc0 = {0.f, 0.f, 0.f, 0.f, 0.f, 0.f, 0.f, 0.f};
    v8f acc1 = {0.f, 0.f, 0.f, 0.f, 0.f, 0.f, 0.f, 0.f};
    #pragma unroll 2
    for (int i = 0; i < 16; ++i) {               // 2 chunks/iter, 32 total
      const int kb0 = (half + (i << 2)) << 5;
      const int kb1 = kb0 + 64;
      __builtin_prefetch((const void*)(vrow + kb0 + 128), 0, 1);
      V16 a0, a1;
      a0.h[0] = *(const h8*)(arow + kb0);        // ds_load_b128: K 0..7 (+cofs)
      a0.h[1] = *(const h8*)(arow + kb0 + 8);    // ds_load_b128: K 8..15 (+cofs)
      a1.h[0] = *(const h8*)(arow + kb1);
      a1.h[1] = *(const h8*)(arow + kb1 + 8);
      const v16h bv0 = *(const v16h*)(vrow + (size_t)kb0);  // 32B contiguous
      const v16h bv1 = *(const v16h*)(vrow + (size_t)kb1);
      acc0 = __builtin_amdgcn_wmma_f32_16x16x32_f16(false, a0.v, false, bv0,
                                                    (short)0, acc0, false, false);
      acc1 = __builtin_amdgcn_wmma_f32_16x16x32_f16(false, a1.v, false, bv1,
                                                    (short)0, acc1, false, false);
    }
    #pragma unroll
    for (int r = 0; r < 8; ++r)
      atomicAdd(&outbuf[(r + (lhalf << 3)) * 68 + d], acc0[r] + acc1[r]);
  }
  __syncthreads();

  // ---------- Phase 4: stage-2 retrieval attention (per row) ----------
  if (tid < 16) {
    const int n = n0 + tid;
    const float* qvr = qvbuf + ((size_t)(b * N_ + n)) * 64 + sh * 8;
    float orow[64];
    #pragma unroll
    for (int dd = 0; dd < 64; ++dd) orow[dd] = outbuf[tid * 68 + dd];
    float lg[8];
    #pragma unroll
    for (int r = 0; r < 8; ++r) {
      float l = 0.f;
      #pragma unroll
      for (int hd = 0; hd < 8; ++hd) {
        float kv = gb[r * 8 + hd];
        #pragma unroll
        for (int hh = 0; hh < 8; ++hh)
          kv = __builtin_fmaf(orow[r * 8 + hh], gw[(r * 8 + hh) * 8 + hd], kv);
        l = __builtin_fmaf(qvr[hd], kv, l);
      }
      lg[r] = l;
    }
    float mx = lg[0];
    #pragma unroll
    for (int r = 1; r < 8; ++r) mx = fmaxf(mx, lg[r]);
    float e[8], ssum = 0.f;
    #pragma unroll
    for (int r = 0; r < 8; ++r) { e[r] = __expf(lg[r] - mx); ssum += e[r]; }
    const float inv = 1.f / ssum;
    const size_t vsb = ((size_t)(b * N_ + n) * S_ + sh) * R_;
    float os[8] = {0.f, 0.f, 0.f, 0.f, 0.f, 0.f, 0.f, 0.f};
    #pragma unroll
    for (int r = 0; r < 8; ++r) {
      const float vs = e[r] * inv;
      vscore_out[vsb + r] = vs;
      #pragma unroll
      for (int hh = 0; hh < 8; ++hh)
        os[hh] = __builtin_fmaf(vs, orow[r * 8 + hh], os[hh]);
    }
    float* ob = obuf + ((size_t)(b * N_ + n)) * 64 + sh * 8;
    #pragma unroll
    for (int hh = 0; hh < 8; ++hh) ob[hh] = os[hh];
  }
}

// ================================================================ kernel 3
// o @ wout + bout -> relu(@dec_w1+b) @ dec_w2 + b -> y [B,N,1]
__global__ __launch_bounds__(256) void k_decode(
    const float* __restrict__ obuf,
    const float* __restrict__ wout, const float* __restrict__ bout,
    const float* __restrict__ dw1, const float* __restrict__ db1,
    const float* __restrict__ dw2, const float* __restrict__ db2,
    float* __restrict__ y_out) {
  const int idx = blockIdx.x * 256 + threadIdx.x;
  const float* orow = obuf + (size_t)idx * 64;
  float o2[64], t1[64];
  gemv64(orow, wout, bout, o2);
  gemv64(o2, dw1, db1, t1);
  float y = db2[0];
  #pragma unroll
  for (int i = 0; i < 64; ++i) y += fmaxf(t1[i], 0.f) * dw2[i];
  y_out[idx] = y;
}

// ================================================================ launch
extern "C" void kernel_launch(void* const* d_in, const int* in_sizes, int n_in,
                              void* d_out, int out_size, void* d_ws,
                              size_t ws_size, hipStream_t stream) {
  (void)in_sizes; (void)n_in; (void)out_size; (void)ws_size;
  const float* x      = (const float*)d_in[0];
  const float* enc_w1 = (const float*)d_in[1];
  const float* enc_b1 = (const float*)d_in[2];
  const float* enc_w2 = (const float*)d_in[3];
  const float* enc_b2 = (const float*)d_in[4];
  const float* wq  = (const float*)d_in[5];
  const float* bq  = (const float*)d_in[6];
  const float* wk  = (const float*)d_in[7];
  const float* bk  = (const float*)d_in[8];
  const float* wv  = (const float*)d_in[9];
  const float* bv  = (const float*)d_in[10];
  const float* wvq = (const float*)d_in[11];
  const float* bvq = (const float*)d_in[12];
  const float* gw  = (const float*)d_in[13];
  const float* gb  = (const float*)d_in[14];
  const float* wout = (const float*)d_in[15];
  const float* bout = (const float*)d_in[16];
  const float* dw1  = (const float*)d_in[17];
  const float* db1  = (const float*)d_in[18];
  const float* dw2  = (const float*)d_in[19];
  const float* db2  = (const float*)d_in[20];

  // workspace layout (bytes)
  char* ws = (char*)d_ws;
  _Float16* q16   = (_Float16*)(ws + 0);          // 2 MB  [B,S,N,8] f16
  _Float16* k16   = (_Float16*)(ws + (2u << 20)); // 2 MB  [B,S,N,8] f16
  _Float16* vt    = (_Float16*)(ws + (4u << 20)); // 2 MB  [B,64,N]  f16
  float*    qvbuf = (float*)(ws + (6u << 20));    // 4 MB  [B,N,64]  f32
  float*    obuf  = (float*)(ws + (10u << 20));   // 4 MB  [B,N,64]  f32

  // d_out layout: y [B,N,1] | attn [B,S,N,N] | v_score [B,N,S,R]
  float* y_out      = (float*)d_out;
  float* attn_out   = (float*)d_out + 16384;
  float* vscore_out = (float*)d_out + 16384 + (size_t)268435456;

  k_encode<<<64, 256, 0, stream>>>(x, enc_w1, enc_b1, enc_w2, enc_b2,
                                   wq, bq, wk, bk, wv, bv, wvq, bvq,
                                   q16, k16, vt, qvbuf);

  const size_t lds_bytes =
      (size_t)16 * PSTR * sizeof(float) + (size_t)16 * PSTRH * sizeof(_Float16);
  k_attn<<<B_ * S_ * (N_ / 16), 256, lds_bytes, stream>>>(
      q16, k16, vt, qvbuf, gw, gb, attn_out, vscore_out, obuf);

  k_decode<<<64, 256, 0, stream>>>(obuf, wout, bout, dw1, db1, dw2, db2,
                                   y_out);
}
